// WindowAttention_9646496547094
// MI455X (gfx1250) — compile-verified, matrix operand order checked by reading, not run on hardware
//
#include <hip/hip_runtime.h>
#include <hip/hip_bf16.h>

typedef __attribute__((ext_vector_type(16))) __bf16 v16bf;
typedef __attribute__((ext_vector_type(8)))  __bf16 v8bf;
typedef __attribute__((ext_vector_type(4)))  __bf16 v4bf;
typedef __attribute__((ext_vector_type(8)))  float  v8f;
typedef __attribute__((ext_vector_type(4)))  float  v4f;

#define K_TOK 49          // tokens per window
#define K_PAD 64          // padded tokens (4 x 16 WMMA tiles)
#define K_DIM 256         // embed dim
#define K_NH  8           // heads
#define K_HD  32          // head dim
#define K_NW  64          // windows per image (mask dim)
#define K_WSTAGE 128      // weight output-channels staged per pass

// LDS layout (bf16 element offsets)
static constexpr int LDS_Q  = 0;                        // 64 x 256  Q (scaled, biased)
static constexpr int LDS_K  = K_PAD * K_DIM;            // 64 x 256  K
static constexpr int LDS_VT = 2 * K_PAD * K_DIM;        // 256 x 64  V transposed (chan-major)
static constexpr int LDS_XO = 3 * K_PAD * K_DIM;        // 64 x 256  X, later attn-out
static constexpr int LDS_W  = 4 * K_PAD * K_DIM;        // 128 x 256 weight staging
static constexpr int LDS_P  = LDS_W + K_WSTAGE * K_DIM; // 8 waves x 16 x 64 softmax P
static constexpr int LDS_ELEMS = LDS_P + 8 * 16 * K_PAD;      // bf16 elements
static constexpr int BM_ELEMS   = K_NH * K_TOK * K_PAD;       // per-window bias+mask f32
static constexpr int BIAS_ELEMS = 768 + 256;                  // qkv_b | proj_b
static constexpr int LDS_BYTES  = LDS_ELEMS * 2 + (BM_ELEMS + BIAS_ELEMS) * 4; // 317440

// workspace layout (bytes)
static constexpr size_t WS_WQ = 0;                            // 768x256 bf16
static constexpr size_t WS_WP = WS_WQ + 768 * 256 * 2;        // 256x256 bf16
static constexpr size_t WS_BM = WS_WP + 256 * 256 * 2;        // [64][8][49][64] f32

// A / B^T fragment loader for v_wmma_*_16x16x32_bf16.
// ISA 16-bit 16x32 layout: lanes 0-15 hold row (lane&15), K in {0..7,16..23};
// lanes 16-31 hold K in {8..15,24..31}. Source is row-major with 32-aligned K base.
__device__ __forceinline__ v16bf ld_frag(const __bf16* p) {
  union { v16bf v; v8bf h[2]; } u;
  u.h[0] = *(const v8bf*)(p);        // K +0..7   (16B ds_load)
  u.h[1] = *(const v8bf*)(p + 16);   // K +16..23 (16B ds_load)
  return u.v;
}

__device__ __forceinline__ v8f bf_wmma(v16bf a, v16bf b, v8f c) {
  return __builtin_amdgcn_wmma_f32_16x16x32_bf16(false, a, false, b, (short)0, c,
                                                 false, false);
}

__device__ __forceinline__ void cvt_store4(__bf16* d, v4f v) {
  v4bf o;
  o.x = (__bf16)v.x; o.y = (__bf16)v.y; o.z = (__bf16)v.z; o.w = (__bf16)v.w;
  *(v4bf*)d = o;                     // single ds_store_b64
}

// K=256 panel GEMM: 8 software-pipelined WMMAs (next B-frag in flight under WMMA)
__device__ __forceinline__ v8f gemm_k256(const v16bf* afr, const __bf16* wrow) {
  v8f acc = {0.f, 0.f, 0.f, 0.f, 0.f, 0.f, 0.f, 0.f};
  v16bf bcur = ld_frag(wrow);
#pragma unroll
  for (int kk = 0; kk < 8; ++kk) {
    v16bf bnext = bcur;
    if (kk < 7) bnext = ld_frag(wrow + (kk + 1) * 32);   // imm offset kk*64+{0,32}B
    acc = bf_wmma(afr[kk], bcur, acc);
    bcur = bnext;
  }
  return acc;
}

// ---------------- pre-pass: bf16 weights + combined bias/mask tables ---------
__global__ __launch_bounds__(256)
void swin_precompute(const float* __restrict__ qkv_w,
                     const float* __restrict__ proj_w,
                     const float* __restrict__ bias_table,
                     const float* __restrict__ mask,
                     __bf16* __restrict__ wq, __bf16* __restrict__ wp,
                     float* __restrict__ bm)
{
  const int idx = blockIdx.x * 256 + threadIdx.x;
  const int stride = gridDim.x * 256;
  for (int i = idx; i < 768 * 256; i += stride) wq[i] = (__bf16)qkv_w[i];
  for (int i = idx; i < 256 * 256; i += stride) wp[i] = (__bf16)proj_w[i];
  // bm[w][h][n][m pad 64]
  for (int i = idx; i < K_NW * K_NH * K_TOK * K_PAD; i += stride) {
    const int m  = i & 63;
    const int t  = i >> 6;            // (w*8 + h)*49 + n
    const int n  = t % K_TOK;
    const int th = t / K_TOK;
    const int hh = th & 7;
    const int w  = th >> 3;
    float v = -3.0e38f;               // padded rows m>=49 -> -inf-like
    if (m < K_TOK) {
      const int mi = m / 7, mj = m - mi * 7;
      const int ni = n / 7, nj = n - ni * 7;
      const int bi = (mi - ni + 6) * 13 + (mj - nj + 6);
      v = bias_table[bi * K_NH + hh] + mask[(w * K_TOK + m) * K_TOK + n];
    }
    bm[i] = v;
  }
}

__global__ __launch_bounds__(256, 1)
void swin_window_attention(const float* __restrict__ x,
                           const float* __restrict__ qkv_b,
                           const float* __restrict__ proj_b,
                           const __bf16* __restrict__ wq,
                           const __bf16* __restrict__ wp,
                           const float* __restrict__ bm,
                           float* __restrict__ out)
{
  extern __shared__ __bf16 smem[];
  __bf16* sQ  = smem + LDS_Q;
  __bf16* sK  = smem + LDS_K;
  __bf16* sVt = smem + LDS_VT;
  __bf16* sXO = smem + LDS_XO;
  __bf16* sW  = smem + LDS_W;
  __bf16* sP  = smem + LDS_P;
  float*  sBM   = (float*)(smem + LDS_ELEMS);  // byte offset LDS_ELEMS*2, 16B aligned
  float*  sBias = sBM + BM_ELEMS;

  const int tid  = threadIdx.x;
  const int wave = tid >> 5;
  const int lane = tid & 31;
  const int lrow = lane & 15;   // row (A) / col (B) / N (D) index
  const int lhi  = lane >> 4;   // K half (operands) / M half (D)

  // ---------------- stage X (fp32 -> bf16, zero-pad rows 49..63) -------------
  const float* xb = x + (size_t)blockIdx.x * K_TOK * K_DIM;
  for (int i = tid; i < K_PAD * K_DIM / 4; i += 256) {
    const int row = i >> 6;            // DIM/4 == 64 chunks per row
    const int c4  = (i & 63) << 2;
    v4f v = {0.f, 0.f, 0.f, 0.f};
    if (row < K_TOK) v = *(const v4f*)(xb + row * K_DIM + c4);
    cvt_store4(sXO + row * K_DIM + c4, v);
  }
  // -------- copy this window's combined bias+mask table (b128 copies) --------
  {
    const v4f* src = (const v4f*)(bm + (size_t)(blockIdx.x & (K_NW - 1)) * BM_ELEMS);
    v4f* dst = (v4f*)sBM;
    for (int i = tid; i < BM_ELEMS / 4; i += 256) dst[i] = src[i];
  }
  // -------- stage GEMM biases (so inner loops never touch global) ------------
  for (int i = tid; i < 768; i += 256) sBias[i] = qkv_b[i];
  sBias[768 + tid] = proj_b[tid];
  __syncthreads();

  // ---------------- QKV GEMM: 6 stages of 128 output channels ----------------
  // Unrolled so the Q/K/V segment (s>>1) folds at compile time: no exec-mask
  // branching in the write-back.
  const float qscale = 0.17677669529663687f;   // 32^-0.5
  const int   mt     = wave & 3;               // fixed M-tile per wave
  const int   nt0    = wave >> 2;              // N-tile phase (stride 2)
  const __bf16* xrow = sXO + (mt * 16 + lrow) * K_DIM + lhi * 8;

  // A-fragments are invariant across all 6 stages: load once.
  v16bf afr[8];
#pragma unroll
  for (int kk = 0; kk < 8; ++kk) afr[kk] = ld_frag(xrow + kk * 32);

#pragma unroll
  for (int s = 0; s < 6; ++s) {
    {  // stage bf16 weights: plain global_load_b128 -> ds_store_b128 copy
      const v8bf* src = (const v8bf*)(wq + (size_t)s * K_WSTAGE * K_DIM);
      v8bf* dst = (v8bf*)sW;
      for (int i = tid; i < K_WSTAGE * K_DIM / 8; i += 256) dst[i] = src[i];
    }
    if (s < 5)  // hint next weight stage into cache (global_prefetch_b8)
      __builtin_prefetch(wq + (size_t)(s + 1) * K_WSTAGE * K_DIM + tid * 128, 0, 0);
    __syncthreads();

    for (int nt = nt0; nt < 8; nt += 2) {
      const __bf16* wrow = sW + (nt * 16 + lrow) * K_DIM + lhi * 8;
      v8f acc = gemm_k256(afr, wrow);
      const int   c  = s * K_WSTAGE + nt * 16 + lrow;   // output channel (N of D)
      const float bb = sBias[c];
      if (s < 2) {                     // Q: scaled+biased, token-major
        __bf16* qst = sQ + (mt * 16 + lhi * 8) * K_DIM + c;
#pragma unroll
        for (int r = 0; r < 8; ++r)
          qst[r * K_DIM] = (__bf16)((acc[r] + bb) * qscale);
      } else if (s < 4) {              // K: token-major
        __bf16* kst = sK + (mt * 16 + lhi * 8) * K_DIM + (c - 256);
#pragma unroll
        for (int r = 0; r < 8; ++r)
          kst[r * K_DIM] = (__bf16)(acc[r] + bb);
      } else {                         // V: transposed, tokens contiguous
        v8bf pv;                       // -> single ds_store_b128
#pragma unroll
        for (int r = 0; r < 8; ++r) pv[r] = (__bf16)(acc[r] + bb);
        *(v8bf*)(sVt + (c - 512) * K_PAD + mt * 16 + lhi * 8) = pv;
      }
    }
    __syncthreads();
  }

  // ---------------- attention: wave w owns head w ----------------------------
  const int h = wave;
  const __bf16* qrow  = sQ + lrow * K_DIM + h * K_HD + lhi * 8;
  const __bf16* krow  = sK + lrow * K_DIM + h * K_HD + lhi * 8;
  const __bf16* vrow  = sVt + ((size_t)h * K_HD + lrow) * K_PAD + lhi * 8;
  __bf16* Pw    = sP + wave * 16 * K_PAD;
  const __bf16* prow  = Pw + lrow * K_PAD + lhi * 8;
  __bf16* pst   = Pw + (lhi * 8) * K_PAD + lrow;
  const float*  bmrow = sBM + h * K_TOK * K_PAD + lrow * K_PAD + lhi * 8;
  __bf16* ost   = sXO + (lhi * 8) * K_DIM + h * K_HD + lrow;

#pragma unroll
  for (int rt = 0; rt < 4; ++rt) {
    // S = Q Kt : head_dim == 32 -> exactly one WMMA per 16x16 tile
    v8f sacc[4];
    {
      v16bf a = ld_frag(qrow + rt * 16 * K_DIM);
#pragma unroll
      for (int ct = 0; ct < 4; ++ct) {
        v16bf b = ld_frag(krow + ct * 16 * K_DIM);
        v8f z = {0.f, 0.f, 0.f, 0.f, 0.f, 0.f, 0.f, 0.f};
        sacc[ct] = bf_wmma(a, b, z);
      }
    }
    // staged bias+mask (2x ds_load_b128 per tile) + softmax in C/D layout
    float rmax[8], rinv[8], rsum[8];
#pragma unroll
    for (int r = 0; r < 8; ++r) rmax[r] = -3.0e38f;
#pragma unroll
    for (int ct = 0; ct < 4; ++ct) {
      const int n = ct * 16 + lrow;
      float bm8[8];
      if (n < K_TOK) {
        const float* bp = bmrow + ct * 16 * K_PAD + rt * 16;   // 8 consecutive m
        const v4f b0 = *(const v4f*)(bp);
        const v4f b1 = *(const v4f*)(bp + 4);
        bm8[0] = b0.x; bm8[1] = b0.y; bm8[2] = b0.z; bm8[3] = b0.w;
        bm8[4] = b1.x; bm8[5] = b1.y; bm8[6] = b1.z; bm8[7] = b1.w;
      } else {
#pragma unroll
        for (int r = 0; r < 8; ++r) bm8[r] = -3.0e38f;
      }
#pragma unroll
      for (int r = 0; r < 8; ++r) {
        const float v = (n < K_TOK) ? (sacc[ct][r] + bm8[r]) : -3.0e38f;
        sacc[ct][r] = v;
        rmax[r] = fmaxf(rmax[r], v);
      }
    }
#pragma unroll
    for (int r = 0; r < 8; ++r) {        // row max across the 16-lane half
      float v = rmax[r];
      v = fmaxf(v, __shfl_xor(v, 1, 32));
      v = fmaxf(v, __shfl_xor(v, 2, 32));
      v = fmaxf(v, __shfl_xor(v, 4, 32));
      v = fmaxf(v, __shfl_xor(v, 8, 32));
      rmax[r] = v;
      rsum[r] = 0.f;
    }
#pragma unroll
    for (int ct = 0; ct < 4; ++ct)
#pragma unroll
      for (int r = 0; r < 8; ++r) {
        const float e = __expf(sacc[ct][r] - rmax[r]);
        sacc[ct][r] = e;
        rsum[r] += e;
      }
#pragma unroll
    for (int r = 0; r < 8; ++r) {        // row sum across the 16-lane half
      float v = rsum[r];
      v += __shfl_xor(v, 1, 32);
      v += __shfl_xor(v, 2, 32);
      v += __shfl_xor(v, 4, 32);
      v += __shfl_xor(v, 8, 32);
      rinv[r] = 1.0f / v;
    }
    // P -> bf16 LDS (wave-private; LDS ops are in-order within a wave)
#pragma unroll
    for (int ct = 0; ct < 4; ++ct)
#pragma unroll
      for (int r = 0; r < 8; ++r)
        pst[r * K_PAD + ct * 16] = (__bf16)(sacc[ct][r] * rinv[r]);

    // O = P V : M=16, N=32, K=64 -> 2x2 WMMAs (padded K cols carry P==0)
    v8f oacc[2];
    oacc[0] = v8f{0.f, 0.f, 0.f, 0.f, 0.f, 0.f, 0.f, 0.f};
    oacc[1] = v8f{0.f, 0.f, 0.f, 0.f, 0.f, 0.f, 0.f, 0.f};
#pragma unroll
    for (int kk = 0; kk < 2; ++kk) {
      v16bf a = ld_frag(prow + kk * 32);
#pragma unroll
      for (int nt = 0; nt < 2; ++nt) {
        v16bf b = ld_frag(vrow + nt * 16 * K_PAD + kk * 32);
        oacc[nt] = bf_wmma(a, b, oacc[nt]);
      }
    }
#pragma unroll
    for (int nt = 0; nt < 2; ++nt)
#pragma unroll
      for (int r = 0; r < 8; ++r)
        ost[(rt * 16 + r) * K_DIM + nt * 16] = (__bf16)oacc[nt][r];
  }
  __syncthreads();   // all heads' attn-out in sXO

  // ---------------- output projection: 2 stages of 128 channels --------------
  // A-fragments (attn-out) invariant across both stages: load once.
  v16bf ofr[8];
#pragma unroll
  for (int kk = 0; kk < 8; ++kk) ofr[kk] = ld_frag(xrow + kk * 32);

#pragma unroll
  for (int s = 0; s < 2; ++s) {
    if (s) __syncthreads();            // stage-0 reads of sW done before restage
    {
      const v8bf* src = (const v8bf*)(wp + (size_t)s * K_WSTAGE * K_DIM);
      v8bf* dst = (v8bf*)sW;
      for (int i = tid; i < K_WSTAGE * K_DIM / 8; i += 256) dst[i] = src[i];
    }
    __syncthreads();

    float* ob = out + (size_t)blockIdx.x * K_TOK * K_DIM;
    for (int nt = nt0; nt < 8; nt += 2) {
      const __bf16* wrow = sW + (nt * 16 + lrow) * K_DIM + lhi * 8;
      v8f acc = gemm_k256(ofr, wrow);
      const int   c  = s * K_WSTAGE + nt * 16 + lrow;
      const float pb = sBias[768 + c];
      float* op = ob + (mt * 16 + lhi * 8) * K_DIM + c;
#pragma unroll
      for (int r = 0; r < 8; ++r) {
        const int m = mt * 16 + lhi * 8 + r;
        if (m < K_TOK) op[r * K_DIM] = acc[r] + pb;
      }
    }
  }
}

extern "C" void kernel_launch(void* const* d_in, const int* in_sizes, int n_in,
                              void* d_out, int out_size, void* d_ws, size_t ws_size,
                              hipStream_t stream) {
  (void)in_sizes; (void)n_in; (void)ws_size; (void)out_size;
  const float* x          = (const float*)d_in[0];
  const float* mask       = (const float*)d_in[1];
  const float* qkv_w      = (const float*)d_in[2];
  const float* qkv_b      = (const float*)d_in[3];
  const float* proj_w     = (const float*)d_in[4];
  const float* proj_b     = (const float*)d_in[5];
  const float* bias_table = (const float*)d_in[6];
  // d_in[7] = rel_index: recomputed arithmetically in the pre-pass.
  float* out = (float*)d_out;

  char* ws = (char*)d_ws;
  __bf16* wq = (__bf16*)(ws + WS_WQ);
  __bf16* wp = (__bf16*)(ws + WS_WP);
  float*  bm = (float*)(ws + WS_BM);

  swin_precompute<<<512, 256, 0, stream>>>(qkv_w, proj_w, bias_table, mask,
                                           wq, wp, bm);

  (void)hipFuncSetAttribute((const void*)swin_window_attention,
                            hipFuncAttributeMaxDynamicSharedMemorySize, LDS_BYTES);
  swin_window_attention<<<4096, 256, LDS_BYTES, stream>>>(
      x, qkv_b, proj_b, wq, wp, bm, out);
}
